// xNNUEModel_12618613916089
// MI455X (gfx1250) — compile-verified, matrix operand order checked by reading, not run on hardware
//
#include <hip/hip_runtime.h>

// ---------------------------------------------------------------------------
// Fused 4-layer quantized MLP for MI455X (gfx1250, wave32, WMMA bf16).
//   h = f(x W1^T + b1); h = f(h W2^T + b2); h = f(h W3^T + b3); y = f(h W4^T + b4)
//   f = ste_quant(leaky_relu(.))  (forward value: 2*s - floor(clip(s/64)))
//
// Roofline: 224 GFLOP total (92% in layer 1), 402 MB mandatory HBM traffic
// (~17 us @ 23.3 TB/s). fp32 WMMA is only 16x16x4 -> use bf16 16x16x32 with
// f32 accumulation (~8x more MACs/instr). All intermediates stay in LDS
// (120 KB/WG, safe even in CU mode); weights are pre-converted to bf16 in
// d_ws (~1.7 MB) and served from the 192 MB L2.
//
// Layer 1 inner loop is register-blocked 2x in N: one LDS A-fragment feeds
// two WMMAs (two B-fragments from L2), halving LDS read traffic per FLOP.
// ---------------------------------------------------------------------------

typedef __attribute__((ext_vector_type(16))) __bf16 v16bf;
typedef __attribute__((ext_vector_type(8)))  __bf16 v8bf;
typedef __attribute__((ext_vector_type(4)))  __bf16 v4bf;
typedef __attribute__((ext_vector_type(8)))  float  v8f;

#define IN_SZ  768
#define H1N    1024
#define H2N    64
#define H3N    32

#define MT     32            // rows per block
// LDS pitches (elements); +8 bf16 pad => consecutive rows differ by 4 banks,
// so the 16 lanes of a 16B A-fragment load hit disjoint banks.
#define XP     (IN_SZ + 8)   // 776
#define H1P    (H1N + 8)     // 1032
#define H2P    (H2N + 8)     // 72
#define H3P    (H3N + 8)     // 40
#define SMEM_BYTES ((MT * (XP + H1P + H2P + H3P)) * 2)   // 122880 B

// bf16 weight workspace layout
#define W1_OFF 0
#define W2_OFF (H1N * IN_SZ)                  // 786432
#define W3_OFF (W2_OFF + H2N * H1N)           // 851968
#define W4_OFF (W3_OFF + H3N * H2N)           // 854016
#define W_TOT  (W4_OFF + 1 * H3N)             // 854048

__device__ __forceinline__ float act_quant(float z, float lo, float hi) {
    const float TWO_PI     = 6.28318530717958647693f;
    const float INV_TWO_PI = 0.15915494309189533577f;
    float a = (z >= 0.0f) ? z : 0.01f * z;                 // leaky relu
    float s = a - __sinf(TWO_PI * a) * INV_TWO_PI;         // smooth_floor x2
    s       = s - __sinf(TWO_PI * s) * INV_TWO_PI;
    float xq = floorf(fminf(fmaxf(s * 0.015625f, lo), hi));
    return 2.0f * s - xq;                                  // x + (x - xq)
}

// 16x32 bf16 fragment (A-layout: lanes 0-15 hold K 0..7,16..23 of row M,
// lanes 16-31 hold K 8..15,24..31). `base` = this lane's row pointer, `off`
// already includes k0 and the lane-half offset. Two 16B loads + shuffle.
__device__ __forceinline__ v16bf frag16(const __bf16* base, int off) {
    v8bf lo = *(const v8bf*)(base + off);
    v8bf hi = *(const v8bf*)(base + off + 16);
    return __builtin_shufflevector(lo, hi, 0, 1, 2, 3, 4, 5, 6, 7,
                                           8, 9, 10, 11, 12, 13, 14, 15);
}

__global__ void cvt_weights_bf16(const float* __restrict__ w1,
                                 const float* __restrict__ w2,
                                 const float* __restrict__ w3,
                                 const float* __restrict__ w4,
                                 __bf16* __restrict__ ws) {
    int i = blockIdx.x * blockDim.x + threadIdx.x;
    if (i >= W_TOT) return;
    float v;
    if (i < W2_OFF)      v = w1[i];
    else if (i < W3_OFF) v = w2[i - W2_OFF];
    else if (i < W4_OFF) v = w3[i - W3_OFF];
    else                 v = w4[i - W4_OFF];
    ws[i] = (__bf16)v;
}

__global__ void __launch_bounds__(256, 2)
nnue_fused(const float* __restrict__ x,
           const float* __restrict__ b1, const float* __restrict__ b2,
           const float* __restrict__ b3, const float* __restrict__ b4,
           const __bf16* __restrict__ wsb,
           float* __restrict__ out) {
    extern __shared__ __bf16 smem[];
    __bf16* XT  = smem;                  // [MT][XP]
    __bf16* H1s = XT  + MT * XP;         // [MT][H1P]
    __bf16* H2s = H1s + MT * H1P;        // [MT][H2P]
    __bf16* H3s = H2s + MT * H2P;        // [MT][H3P]

    const __bf16* w1b = wsb + W1_OFF;
    const __bf16* w2b = wsb + W2_OFF;
    const __bf16* w3b = wsb + W3_OFF;
    const __bf16* w4b = wsb + W4_OFF;

    const int tid  = threadIdx.x;
    const int lane = tid & 31;
    const int wv   = tid >> 5;
    const int l15  = lane & 15;
    const int mhalf = (lane >> 4) << 3;          // C/D layout: M = v + 8*(lane>=16)
    const int aoff  = (lane & 16) ? 8 : 0;       // A/B fragment lane-half K offset
    const size_t row0 = (size_t)blockIdx.x * MT;

    // ---- stage x tile -> LDS bf16 (coalesced float4 loads; x = 402 MB) -----
    {
        const float4* xg = (const float4*)(x + row0 * IN_SZ);
        #pragma unroll
        for (int i = 0; i < (MT * IN_SZ / 4) / 256; ++i) {   // 24 iters
            int e4 = tid + i * 256;
            float4 f = xg[e4];
            int be = e4 * 4;
            int r  = be / IN_SZ;
            int c  = be % IN_SZ;                 // IN_SZ % 4 == 0 -> same row
            v4bf p;
            p[0] = (__bf16)f.x; p[1] = (__bf16)f.y;
            p[2] = (__bf16)f.z; p[3] = (__bf16)f.w;
            *(v4bf*)(&XT[r * XP + c]) = p;
        }
    }
    __syncthreads();

    // ---- layer 1: [32x768] x [768x1024] -> H1s, 2x N register blocking -----
    {
        const int r0 = (wv >> 2) * 16;           // 2 row slices
        const int cbase = (wv & 3) * 256;        // 4 col slices of 256 outputs
        const __bf16* arow = XT + (r0 + l15) * XP;
        for (int chp = 0; chp < 8; ++chp) {      // 8 chunk-pairs
            int n0 = cbase + chp * 32 + l15;
            int n1 = n0 + 16;
            const __bf16* brow0 = w1b + (size_t)n0 * IN_SZ;
            const __bf16* brow1 = w1b + (size_t)n1 * IN_SZ;
            float bias0 = b1[n0];
            float bias1 = b1[n1];
            v8f c0 = {};
            v8f c1 = {};
            for (int ks = 0; ks < IN_SZ / 32; ++ks) {        // 24 k-steps
                int ko = ks * 32 + aoff;
                v16bf af  = frag16(arow, ko);                // 1 LDS A-frag ...
                v16bf bf0 = frag16(brow0, ko);
                v16bf bf1 = frag16(brow1, ko);
                c0 = __builtin_amdgcn_wmma_f32_16x16x32_bf16(
                        false, af, false, bf0, (short)0, c0, false, false);
                c1 = __builtin_amdgcn_wmma_f32_16x16x32_bf16(
                        false, af, false, bf1, (short)0, c1, false, false);
            }                                                // ... feeds 2 WMMAs
            #pragma unroll
            for (int v = 0; v < 8; ++v) {
                int m = r0 + mhalf + v;
                H1s[m * H1P + n0] = (__bf16)act_quant(c0[v] + bias0, -128.0f, 127.0f);
                H1s[m * H1P + n1] = (__bf16)act_quant(c1[v] + bias1, -128.0f, 127.0f);
            }
        }
    }
    __syncthreads();

    // ---- layer 2: [32x1024] x [1024x64] -> H2s ------------------------------
    {
        const int r0 = (wv >> 2) * 16;
        int n = (wv & 3) * 16 + l15;             // 4 col slices of 16 outputs
        const __bf16* arow = H1s + (r0 + l15) * H1P;
        const __bf16* brow = w2b + (size_t)n * H1N;
        float bias = b2[n];
        v8f c = {};
        for (int ks = 0; ks < H1N / 32; ++ks) {              // 32 WMMAs
            int ko = ks * 32 + aoff;
            v16bf af = frag16(arow, ko);
            v16bf bf = frag16(brow, ko);
            c = __builtin_amdgcn_wmma_f32_16x16x32_bf16(
                    false, af, false, bf, (short)0, c, false, false);
        }
        #pragma unroll
        for (int v = 0; v < 8; ++v) {
            float r = act_quant(c[v] + bias, -128.0f, 127.0f);
            H2s[(r0 + mhalf + v) * H2P + n] = (__bf16)r;
        }
    }
    __syncthreads();

    // ---- layer 3: [32x64] x [64x32] -> H3s (waves 0..3) ---------------------
    if (wv < 4) {
        const int r0 = (wv >> 1) * 16;
        int n = (wv & 1) * 16 + l15;
        const __bf16* arow = H2s + (r0 + l15) * H2P;
        const __bf16* brow = w3b + (size_t)n * H2N;
        float bias = b3[n];
        v8f c = {};
        for (int ks = 0; ks < H2N / 32; ++ks) {              // 2 WMMAs
            int ko = ks * 32 + aoff;
            v16bf af = frag16(arow, ko);
            v16bf bf = frag16(brow, ko);
            c = __builtin_amdgcn_wmma_f32_16x16x32_bf16(
                    false, af, false, bf, (short)0, c, false, false);
        }
        #pragma unroll
        for (int v = 0; v < 8; ++v) {
            float r = act_quant(c[v] + bias, -128.0f, 127.0f);
            H3s[(r0 + mhalf + v) * H3P + n] = (__bf16)r;
        }
    }
    __syncthreads();

    // ---- layer 4: [32x32] x [32x1] -> out (VALU dot, one thread per row) ----
    if (tid < MT) {
        float acc = b4[0];
        #pragma unroll
        for (int j = 0; j < H3N; ++j)
            acc += (float)H3s[tid * H3P + j] * (float)w4b[j];
        out[row0 + tid] = act_quant(acc, -2147483648.0f, 2147483647.0f);
    }
}

extern "C" void kernel_launch(void* const* d_in, const int* in_sizes, int n_in,
                              void* d_out, int out_size, void* d_ws, size_t ws_size,
                              hipStream_t stream) {
    const float* x  = (const float*)d_in[0];
    const float* w1 = (const float*)d_in[1];
    const float* b1 = (const float*)d_in[2];
    const float* w2 = (const float*)d_in[3];
    const float* b2 = (const float*)d_in[4];
    const float* w3 = (const float*)d_in[5];
    const float* b3 = (const float*)d_in[6];
    const float* w4 = (const float*)d_in[7];
    const float* b4 = (const float*)d_in[8];
    __bf16* wsb = (__bf16*)d_ws;

    // 1) fp32 -> bf16 weight conversion into workspace (~1.7 MB, L2-resident)
    cvt_weights_bf16<<<(W_TOT + 255) / 256, 256, 0, stream>>>(w1, w2, w3, w4, wsb);

    // 2) fused 4-layer WMMA pipeline, 32 rows/block, 120 KB dynamic LDS
    (void)hipFuncSetAttribute((const void*)nnue_fused,
                              hipFuncAttributeMaxDynamicSharedMemorySize,
                              SMEM_BYTES);
    int rows = in_sizes[0] / IN_SZ;
    nnue_fused<<<rows / MT, 256, SMEM_BYTES, stream>>>(x, b1, b2, b3, b4, wsb,
                                                       (float*)d_out);
}
